// DynamicTokenSelector_61435212202335
// MI455X (gfx1250) — matrix operand order repaired and checked by reading, not compiled, not run on hardware
//
#include <hip/hip_runtime.h>
#include <hip/hip_bf16.h>
#include <math.h>

// ---------------- problem constants (from reference) ----------------
#define B_    16
#define N_    4096
#define D_    1024
#define H_    256
#define KEEP_ 2867          // max(1, int(4096 * 0.7))

typedef __attribute__((ext_vector_type(16))) __bf16 v16bf;
typedef __attribute__((ext_vector_type(8)))  float  v8f;

// RNE float -> bf16 bits
__device__ __forceinline__ unsigned short f32_to_bf16_rne(float v) {
    unsigned int u = __float_as_uint(v);
    unsigned int r = (u + 0x7FFFu + ((u >> 16) & 1u)) >> 16;
    return (unsigned short)r;
}

// --------------------------------------------------------------------
// Kernel 1: pre-swizzle W1 (f32 [D][H] row-major) into bf16 B-fragment
// register order so the score kernel loads one 32B chunk per lane.
// Layout: ushort wsW[htile(16)][kstep(32)][lane(32)][e(16)]
//   lane <  16 : column h = htile*16 + lane,      K = kstep*32 + e
//   lane >= 16 : column h = htile*16 + lane - 16, K = kstep*32 + 16 + e
// --------------------------------------------------------------------
__global__ void prep_w1_bf16(const float* __restrict__ W1,
                             unsigned short* __restrict__ wsW) {
    int gid  = blockIdx.x * blockDim.x + threadIdx.x;   // 0 .. 262143
    int htile = gid >> 14;                              // 32*32*16 = 16384 per tile
    int rem   = gid & 16383;
    int kstep = rem >> 9;                               // 32*16 = 512 per kstep
    int rem2  = rem & 511;
    int lane  = rem2 >> 4;
    int e     = rem2 & 15;
    int h = htile * 16 + (lane & 15);
    int k = kstep * 32 + ((lane < 16) ? e : (16 + e));
    wsW[gid] = f32_to_bf16_rne(W1[k * H_ + h]);
}

// --------------------------------------------------------------------
// Kernel 2: scorer GEMM + GELU + dot(W2) + sigmoid, via bf16 WMMA.
// Block = 512 threads = 16 waves; block handles 16 tokens.
// All 16 waves share the SAME A matrix, so the block stages the whole
// 16x1024 x-tile ONCE into LDS as bf16, pre-swizzled into per-lane
// A-fragment order (one aligned 32B ds_load per lane per k-step).
// Each wave then owns hidden columns [16w,16w+16) and streams only its
// B fragments from global (512 KB total -> L2-resident, reused by all
// 4096 blocks).  C accumulates in f32 over K=1024 (32 WMMA ops/wave).
// --------------------------------------------------------------------
__global__ void __launch_bounds__(512)
score_kernel(const float* __restrict__ x,
             const unsigned short* __restrict__ wsW,
             const float* __restrict__ b1,
             const float* __restrict__ W2,
             const float* __restrict__ b2,
             float* __restrict__ scores) {
    // A fragments in register order: aFrag[kstep(32)][lane(32)] = 32 KB
    __shared__ v16bf aFrag[32 * 32];
    __shared__ float red[16];

    const int tid  = threadIdx.x;
    const int wave = tid >> 5;          // 0..15 == htile
    const int lane = tid & 31;
    const int gtok0 = blockIdx.x * 16;  // 16 tokens per block (same batch: N%16==0)

    if (tid < 16) red[tid] = 0.0f;

    // ---- stage phase: convert x-tile (f32) -> bf16 A fragments in LDS.
    // Pair p = (kstep, lane); 1024 pairs, each thread fills 2.
    // A 16x32 layout used here (contiguous k, same as packed B; not ISA table order): row M = lane&15 for both halves;
    // lanes<16 take K {0..15}, lanes>=16 take K {16..31}.
    for (int p = tid; p < 1024; p += 512) {
        const int ks    = p >> 5;
        const int flane = p & 31;
        const int row   = flane & 15;
        const int koff  = (flane < 16) ? 0 : 16;
        const float* xr = x + (size_t)(gtok0 + row) * D_ + ks * 32 + koff;
        float4 f0 = *(const float4*)(xr);
        float4 f1 = *(const float4*)(xr + 4);
        float4 f2 = *(const float4*)(xr + 8);
        float4 f3 = *(const float4*)(xr + 12);
        v16bf a;
        a[0]  = (__bf16)f0.x;  a[1]  = (__bf16)f0.y;
        a[2]  = (__bf16)f0.z;  a[3]  = (__bf16)f0.w;
        a[4]  = (__bf16)f1.x;  a[5]  = (__bf16)f1.y;
        a[6]  = (__bf16)f1.z;  a[7]  = (__bf16)f1.w;
        a[8]  = (__bf16)f2.x;  a[9]  = (__bf16)f2.y;
        a[10] = (__bf16)f2.z;  a[11] = (__bf16)f2.w;
        a[12] = (__bf16)f3.x;  a[13] = (__bf16)f3.y;
        a[14] = (__bf16)f3.z;  a[15] = (__bf16)f3.w;
        aFrag[p] = a;
    }
    __syncthreads();

    const int hcol = wave * 16 + (lane & 15);
    const float b1v = b1[hcol];
    const float w2v = W2[hcol];

    const v16bf* bbase = (const v16bf*)wsW + (size_t)wave * 32 * 32; // [kstep][lane]

    // ---- compute phase: A from LDS, B from global (L2), WMMA accumulate.
    v8f c = {};
    #pragma unroll 4
    for (int ks = 0; ks < 32; ++ks) {
        v16bf a     = aFrag[ks * 32 + lane];   // 32B ds_load, conflict-free
        v16bf bfrag = bbase[ks * 32 + lane];   // 32B contiguous per lane
        c = __builtin_amdgcn_wmma_f32_16x16x32_bf16(
                false, a, false, bfrag, (short)0, c, false, false);
    }

    // Epilogue: C layout — VGPR r, lane l: M = r + 8*(l>=16), Ncol = l&15.
    #pragma unroll
    for (int r = 0; r < 8; ++r) {
        float v = c[r] + b1v;
        float g = 0.5f * v * (1.0f + erff(v * 0.70710678118654752f)); // exact GELU
        float contrib = g * w2v;
        // sum the 16 hidden columns held by this wave's lane group
        contrib += __shfl_xor(contrib, 1, 32);
        contrib += __shfl_xor(contrib, 2, 32);
        contrib += __shfl_xor(contrib, 4, 32);
        contrib += __shfl_xor(contrib, 8, 32);
        if ((lane & 15) == 0) {
            int m = r + ((lane >= 16) ? 8 : 0);
            atomicAdd(&red[m], contrib);       // ds_add_f32 across 16 waves
        }
    }
    __syncthreads();

    if (tid < 16) {
        float s = red[tid] + b2[0];
        scores[gtok0 + tid] = 1.0f / (1.0f + expf(-s));   // sigmoid
    }
}

// --------------------------------------------------------------------
// Kernel 3: exact per-batch top-k via LDS bitonic sort of 64-bit keys.
// Key = (orderable(score) << 32) | (N-1-index)  sorted descending
//   -> ties resolved to the smaller index, matching jax.lax.top_k.
// --------------------------------------------------------------------
__global__ void __launch_bounds__(1024)
topk_sort_kernel(const float* __restrict__ scores, int* __restrict__ idxOut) {
    __shared__ unsigned long long keys[N_];   // 32 KB of the 320 KB LDS
    const int b   = blockIdx.x;
    const int tid = threadIdx.x;
    const float* s = scores + (size_t)b * N_;

    for (int i = tid; i < N_; i += 1024) {
        unsigned int u = __float_as_uint(s[i]);
        unsigned int key = (u & 0x80000000u) ? ~u : (u | 0x80000000u);
        keys[i] = ((unsigned long long)key << 32) | (unsigned int)(N_ - 1 - i);
    }
    __syncthreads();

    for (int k = 2; k <= N_; k <<= 1) {
        for (int j = k >> 1; j > 0; j >>= 1) {
            for (int i = tid; i < N_; i += 1024) {
                int ixj = i ^ j;
                if (ixj > i) {
                    bool desc = ((i & k) == 0);
                    unsigned long long a  = keys[i];
                    unsigned long long bb = keys[ixj];
                    if ((a < bb) == desc) { keys[i] = bb; keys[ixj] = a; }
                }
            }
            __syncthreads();
        }
    }

    for (int i = tid; i < KEEP_; i += 1024)
        idxOut[(size_t)b * KEEP_ + i] = N_ - 1 - (int)(keys[i] & 0xFFFFFFFFu);
}

// --------------------------------------------------------------------
// Kernel 4: gather kept rows of x (pure float4 streaming, HBM-bound).
// --------------------------------------------------------------------
__global__ void gather_kernel(const float* __restrict__ x,
                              const int* __restrict__ idx,
                              float* __restrict__ kept) {
    const int rowId = blockIdx.x;              // 0 .. B*KEEP-1
    const int b = rowId / KEEP_;
    const int j = rowId - b * KEEP_;
    const int src = idx[(size_t)b * KEEP_ + j];
    const float4* in  = (const float4*)(x + ((size_t)b * N_ + src) * D_);
    float4*       out = (float4*)(kept + (size_t)rowId * D_);
    #pragma unroll 2
    for (int t = threadIdx.x; t < D_ / 4; t += blockDim.x) out[t] = in[t];
}

// --------------------------------------------------------------------
extern "C" void kernel_launch(void* const* d_in, const int* in_sizes, int n_in,
                              void* d_out, int out_size, void* d_ws, size_t ws_size,
                              hipStream_t stream) {
    const float* x  = (const float*)d_in[0];   // [B,N,D] f32
    const float* W1 = (const float*)d_in[1];   // [D,H]   f32
    const float* b1 = (const float*)d_in[2];   // [H]     f32
    const float* W2 = (const float*)d_in[3];   // [H,1]   f32
    const float* b2 = (const float*)d_in[4];   // [1]     f32

    float* outF   = (float*)d_out;
    float* kept   = outF;                                     // [B,KEEP,D] f32
    int*   idxOut = (int*)(outF + (size_t)B_ * KEEP_ * D_);   // [B,KEEP] i32 (bit-cast tail)

    // workspace: [0, 512KB) packed bf16 W1 fragments; then scores [B*N] f32
    unsigned short* wsW   = (unsigned short*)d_ws;
    float*          wsScr = (float*)((char*)d_ws + (size_t)16 * 32 * 32 * 16 * sizeof(unsigned short));

    hipLaunchKernelGGL(prep_w1_bf16, dim3(1024), dim3(256), 0, stream, W1, wsW);
    hipLaunchKernelGGL(score_kernel, dim3((B_ * N_) / 16), dim3(512), 0, stream,
                       x, wsW, b1, W2, b2, wsScr);
    hipLaunchKernelGGL(topk_sort_kernel, dim3(B_), dim3(1024), 0, stream, wsScr, idxOut);
    hipLaunchKernelGGL(gather_kernel, dim3(B_ * KEEP_), dim3(256), 0, stream, x, idxOut, kept);
}
